// ST_Transformer_28406913695894
// MI455X (gfx1250) — matrix operand order repaired and checked by reading, not compiled, not run on hardware
//
#include <hip/hip_runtime.h>

// ---------------------------------------------------------------------------
// CDNA5 (gfx1250, wave32) ST-Transformer. GEMMs: v_wmma_f32_16x16x32_f16 with
// double-buffered LDS tiles filled by global_load_async_to_lds_b128 (ASYNCcnt),
// all operands pre-converted to f16 (K zero-padded to multiples of 32).
// Each wave computes a 32x32 output tile (4 WMMAs per k-step).
// ---------------------------------------------------------------------------

typedef __attribute__((ext_vector_type(16))) _Float16 v16h;
typedef __attribute__((ext_vector_type(8)))  _Float16 v8h;
typedef __attribute__((ext_vector_type(8)))  float    v8f;

#define NROWS 12288L   // B*T = 1024*12
#define BATCH 1024
#define TT    12

// ---------------- GEMM: C[N,M] = act(A16[N,Kpad] @ W16T[M,Kpad]^T + bias (+res))
// Block tile 128x64: 8 waves (4 row x 2 col), each wave = 32 rows x 32 cols
// (2x2 grid of 16x16 WMMA tiles). K step 32, no k guards.
#define BM 128
#define BN 64
#define KT 32
#define LDT 40   // halfs per LDS row (80B: 16B-aligned, bank-skewed)

__global__ __launch_bounds__(256) void gemm_f16_wmma(
    const _Float16* __restrict__ A16, int lda,
    const _Float16* __restrict__ W16T, int kpad,
    const float* __restrict__ bias, const float* __restrict__ res,
    float* __restrict__ C, _Float16* __restrict__ C16,
    int M, int relu)
{
    __shared__ __align__(16) _Float16 As[2][BM * LDT];
    __shared__ __align__(16) _Float16 Bs[2][BN * LDT];

    const int tid  = threadIdx.x;
    const int row0 = blockIdx.y * BM;
    const int col0 = blockIdx.x * BN;
    const int w    = tid >> 5;          // wave 0..7
    const int lane = tid & 31;
    const int wr   = w >> 1;            // 0..3 (32-row slabs)
    const int wc   = w & 1;             // 0..1 (32-col slabs)
    const int h    = lane >> 4;
    const int r    = lane & 15;

    // staging coordinates: thread covers 16B; A tile needs two passes (128 rows)
    const int srow = tid >> 2;          // 0..63
    const int skk  = (tid & 3) * 8;     // 0,8,16,24
    const long aBase0 = (long)(row0 + srow) * lda + skk;        // rows 0..63
    const long aBase1 = (long)(row0 + 64 + srow) * lda + skk;   // rows 64..127
    const int  gcB    = col0 + srow;
    const long bBase  = (long)gcB * kpad + skk;
    const bool bOk    = (gcB < M);

    const unsigned ldsA0a = (unsigned)(uintptr_t)&As[0][srow * LDT + skk];
    const unsigned ldsA0b = (unsigned)(uintptr_t)&As[0][(64 + srow) * LDT + skk];
    const unsigned ldsA1a = (unsigned)(uintptr_t)&As[1][srow * LDT + skk];
    const unsigned ldsA1b = (unsigned)(uintptr_t)&As[1][(64 + srow) * LDT + skk];
    const unsigned ldsB0  = (unsigned)(uintptr_t)&Bs[0][srow * LDT + skk];
    const unsigned ldsB1  = (unsigned)(uintptr_t)&Bs[1][srow * LDT + skk];

    // out-of-range B columns: zero once (invariant over k); async path skips them
    if (!bOk) {
        v8h z = {};
        *(v8h*)&Bs[0][srow * LDT + skk] = z;
        *(v8h*)&Bs[1][srow * LDT + skk] = z;
    }

    // async global -> LDS tile copy (ASYNCcnt tracked, no VGPR round-trip)
    auto issue = [&](int ks, int buf) {
        const _Float16* ga0 = &A16[aBase0 + (long)ks * KT];
        const _Float16* ga1 = &A16[aBase1 + (long)ks * KT];
        const unsigned la0 = buf ? ldsA1a : ldsA0a;
        const unsigned la1 = buf ? ldsA1b : ldsA0b;
        asm volatile("global_load_async_to_lds_b128 %0, %1, off"
                     :: "v"(la0), "v"(ga0) : "memory");
        asm volatile("global_load_async_to_lds_b128 %0, %1, off"
                     :: "v"(la1), "v"(ga1) : "memory");
        if (bOk) {
            const _Float16* gb = &W16T[bBase + (long)ks * KT];
            const unsigned lb = buf ? ldsB1 : ldsB0;
            asm volatile("global_load_async_to_lds_b128 %0, %1, off"
                         :: "v"(lb), "v"(gb) : "memory");
        }
    };

    v8f acc00 = {}, acc01 = {}, acc10 = {}, acc11 = {};
    const int ksteps = kpad / KT;

    issue(0, 0);
    for (int ks = 0; ks < ksteps; ++ks) {
        const int cur = ks & 1;
        asm volatile("s_wait_asynccnt 0x0" ::: "memory");
        __syncthreads();                       // tile `cur` complete for all waves
        if (ks + 1 < ksteps) issue(ks + 1, cur ^ 1);  // overlap next copy w/ compute

        // ---- A fragments: rows wr*32+r and wr*32+16+r; K = h*8+[0..7], 16+h*8+[0..7]
        v16h af0, af1, bf0, bf1;
        {
            const _Float16* p = &As[cur][(wr * 32 + r) * LDT];
            v8h lo = *(const v8h*)&p[h * 8];
            v8h hi = *(const v8h*)&p[16 + h * 8];
            #pragma unroll
            for (int j = 0; j < 8; ++j) { af0[j] = lo[j]; af0[8 + j] = hi[j]; }
        }
        {
            const _Float16* p = &As[cur][(wr * 32 + 16 + r) * LDT];
            v8h lo = *(const v8h*)&p[h * 8];
            v8h hi = *(const v8h*)&p[16 + h * 8];
            #pragma unroll
            for (int j = 0; j < 8; ++j) { af1[j] = lo[j]; af1[8 + j] = hi[j]; }
        }
        // ---- B fragments: cols wc*32+r and wc*32+16+r; K = h*8+[0..7], 16+h*8+[0..7] ----
        {
            const _Float16* p = &Bs[cur][(wc * 32 + r) * LDT + h * 8];
            v8h lo = *(const v8h*)&p[0];
            v8h hi = *(const v8h*)&p[16];
            #pragma unroll
            for (int j = 0; j < 8; ++j) { bf0[j] = lo[j]; bf0[8 + j] = hi[j]; }
        }
        {
            const _Float16* p = &Bs[cur][(wc * 32 + 16 + r) * LDT + h * 8];
            v8h lo = *(const v8h*)&p[0];
            v8h hi = *(const v8h*)&p[16];
            #pragma unroll
            for (int j = 0; j < 8; ++j) { bf1[j] = lo[j]; bf1[8 + j] = hi[j]; }
        }
        acc00 = __builtin_amdgcn_wmma_f32_16x16x32_f16(false, af0, false, bf0,
                                                       (short)0, acc00, false, false);
        acc01 = __builtin_amdgcn_wmma_f32_16x16x32_f16(false, af0, false, bf1,
                                                       (short)0, acc01, false, false);
        acc10 = __builtin_amdgcn_wmma_f32_16x16x32_f16(false, af1, false, bf0,
                                                       (short)0, acc10, false, false);
        acc11 = __builtin_amdgcn_wmma_f32_16x16x32_f16(false, af1, false, bf1,
                                                       (short)0, acc11, false, false);
    }

    // ---- epilogue: 2x2 tiles; lane holds col r (+16), rows j + 8*h ----
    #pragma unroll
    for (int ta = 0; ta < 2; ++ta) {
        const long rowb = (long)(row0 + wr * 32 + ta * 16 + 8 * h);
        #pragma unroll
        for (int tb = 0; tb < 2; ++tb) {
            const int gcol = col0 + wc * 32 + tb * 16 + r;
            if (gcol >= M) continue;
            const float bv = bias ? bias[gcol] : 0.0f;
            const v8f acc = ta ? (tb ? acc11 : acc10) : (tb ? acc01 : acc00);
            #pragma unroll
            for (int j = 0; j < 8; ++j) {
                const long gr = rowb + j;
                float v = acc[j] + bv;
                if (res)  v += res[gr * M + gcol];
                if (relu) v = v > 0.0f ? v : 0.0f;
                if (C)    C[gr * M + gcol] = v;
                if (C16)  C16[gr * M + gcol] = (_Float16)v;
            }
        }
    }
}

// ---------------- weight convert: W(K,M) f32 -> WT(M,Kpad) f16, zero pad ----
__global__ void convert_w(const float* __restrict__ W, _Float16* __restrict__ WT,
                          int K, int M, int Kpad)
{
    const long i = (long)blockIdx.x * blockDim.x + threadIdx.x;
    if (i >= (long)M * Kpad) return;
    const int k = (int)(i % Kpad);
    const long m = i / Kpad;
    WT[i] = (k < K) ? (_Float16)W[(long)k * M + m] : (_Float16)0.0f;
}

// ---------------- activation convert: (N,75) f32 -> (N,96) f16, zero pad ----
__global__ void convert_a96(const float* __restrict__ A, _Float16* __restrict__ A16)
{
    const long i = (long)blockIdx.x * blockDim.x + threadIdx.x;
    if (i >= NROWS * 96L) return;
    const int f = (int)(i % 96);
    const long bt = i / 96;
    A16[i] = (f < 75) ? (_Float16)A[bt * 75 + f] : (_Float16)0.0f;
}

// ---------------- LayerNorm (per-row workgroup), writes f32 + f16 -----------
__global__ __launch_bounds__(256) void ln_kernel(
    const float* __restrict__ a, const float* __restrict__ g,
    const float* __restrict__ be, float* __restrict__ out,
    _Float16* __restrict__ out16, int d)
{
    __shared__ float sb[256];
    const long row = blockIdx.x;
    const float* pa = a + row * (long)d;
    float s = 0.0f;
    for (int i = threadIdx.x; i < d; i += 256) s += pa[i];
    sb[threadIdx.x] = s; __syncthreads();
    for (int o = 128; o > 0; o >>= 1) {
        if (threadIdx.x < o) sb[threadIdx.x] += sb[threadIdx.x + o];
        __syncthreads();
    }
    const float m = sb[0] / d; __syncthreads();
    float v = 0.0f;
    for (int i = threadIdx.x; i < d; i += 256) { float t = pa[i] - m; v += t * t; }
    sb[threadIdx.x] = v; __syncthreads();
    for (int o = 128; o > 0; o >>= 1) {
        if (threadIdx.x < o) sb[threadIdx.x] += sb[threadIdx.x + o];
        __syncthreads();
    }
    const float rstd = rsqrtf(sb[0] / d + 1e-5f);
    for (int i = threadIdx.x; i < d; i += 256) {
        const float y = (pa[i] - m) * rstd * g[i] + be[i];
        out[row * (long)d + i] = y;
        if (out16) out16[row * (long)d + i] = (_Float16)y;
    }
}

// ---------------- Attention: one workgroup per (batch, head), T=12 ----------
__global__ __launch_bounds__(256) void attn_kernel(
    const float* __restrict__ Qb, const float* __restrict__ Kb,
    const float* __restrict__ Vb, _Float16* __restrict__ Ob16, int d, int hd)
{
    __shared__ float q[TT * 128], k[TT * 128], v[TT * 128];
    __shared__ float aw[TT * TT];
    const int nh   = d / hd;
    const int head = blockIdx.x % nh;
    const long b   = blockIdx.x / nh;
    const int tid  = threadIdx.x;

    for (int idx = tid; idx < TT * hd; idx += 256) {
        const int i = idx / hd, e = idx % hd;
        const long off = (b * TT + i) * (long)d + head * hd + e;
        q[i * hd + e] = Qb[off];
        k[i * hd + e] = Kb[off];
        v[i * hd + e] = Vb[off];
    }
    __syncthreads();

    const float scale = rsqrtf((float)hd);
    if (tid < TT * TT) {
        const int i = tid / TT, j = tid % TT;
        float s = 0.0f;
        for (int e = 0; e < hd; ++e) s += q[i * hd + e] * k[j * hd + e];
        aw[tid] = s * scale;
    }
    __syncthreads();

    if (tid < TT) {
        float mx = -3.0e38f;
        for (int j = 0; j < TT; ++j) mx = fmaxf(mx, aw[tid * TT + j]);
        float sum = 0.0f;
        for (int j = 0; j < TT; ++j) { float e = __expf(aw[tid * TT + j] - mx); aw[tid * TT + j] = e; sum += e; }
        const float inv = 1.0f / sum;
        for (int j = 0; j < TT; ++j) aw[tid * TT + j] *= inv;
    }
    __syncthreads();

    for (int idx = tid; idx < TT * hd; idx += 256) {
        const int i = idx / hd, e = idx % hd;
        float s = 0.0f;
        for (int j = 0; j < TT; ++j) s += aw[i * TT + j] * v[j * hd + e];
        Ob16[(b * TT + i) * (long)d + head * hd + e] = (_Float16)s;
    }
}

// ---------------- GAT spatial embedding (per (b,t) workgroup) ---------------
__global__ __launch_bounds__(256) void gat_kernel(
    const float* __restrict__ x, const int* __restrict__ adj,
    const float* __restrict__ sW, const float* __restrict__ sa,
    float* __restrict__ sp)
{
    __shared__ float xr[75];
    __shared__ float Wh[25 * 256];
    __shared__ float wh1[25];
    __shared__ float att[25 * 25];
    __shared__ float wbar[25];
    const long bt = blockIdx.x;
    const int tid = threadIdx.x;

    if (tid < 75) xr[tid] = x[bt * 75 + tid];
    __syncthreads();

    for (int idx = tid; idx < 25 * 256; idx += 256) {
        const int j = idx >> 8, hh = idx & 255;
        Wh[idx] = xr[j * 3 + 0] * sW[hh] + xr[j * 3 + 1] * sW[256 + hh]
                + xr[j * 3 + 2] * sW[512 + hh];
    }
    __syncthreads();
    {
        const int w = tid >> 5, lane = tid & 31;
        for (int j = w; j < 25; j += 8) {
            float s = 0.0f;
            for (int e = lane; e < 256; e += 32) s += Wh[j * 256 + e] * sa[e];
            for (int o = 16; o > 0; o >>= 1) s += __shfl_down(s, o, 32);
            if (lane == 0) wh1[j] = s;
        }
    }
    __syncthreads();
    if (tid < 25) {
        float vals[25]; float mx = -3.0e38f;
        for (int kk = 0; kk < 25; ++kk) {
            float e = wh1[tid] + wh1[kk];
            e = e > 0.0f ? e : 0.0f;
            if (adj[tid * 25 + kk] <= 0) e = 0.0f;
            vals[kk] = e; mx = fmaxf(mx, e);
        }
        float sum = 0.0f;
        for (int kk = 0; kk < 25; ++kk) { float e = __expf(vals[kk] - mx); vals[kk] = e; sum += e; }
        const float inv = 1.0f / sum;
        for (int kk = 0; kk < 25; ++kk) att[tid * 25 + kk] = vals[kk] * inv;
    }
    __syncthreads();
    if (tid < 25) {
        float s = 0.0f;
        for (int j = 0; j < 25; ++j) s += att[j * 25 + tid];
        wbar[tid] = s * (1.0f / 25.0f);
    }
    __syncthreads();
    if (tid < 256) {
        float s = 0.0f;
        for (int kk = 0; kk < 25; ++kk) s += wbar[kk] * Wh[kk * 256 + tid];
        sp[bt * 256 + tid] = s;
    }
}

// ---------------- temporal branch: writes f16 (N,96), zero-padded -----------
__global__ void temporal_kernel(const float* __restrict__ x,
                                const float* __restrict__ tW,
                                const float* __restrict__ tb,
                                _Float16* __restrict__ out16)
{
    const long i = (long)blockIdx.x * blockDim.x + threadIdx.x;
    if (i >= NROWS * 96L) return;
    const int f = (int)(i % 96);
    const long bt = i / 96;
    if (f >= 75) { out16[i] = (_Float16)0.0f; return; }
    const int t = (int)(bt % TT);
    const long b = bt / TT;
    float s = tb[t] + x[bt * 75 + f];
    for (int tp = 0; tp < TT; ++tp) s += x[(b * TT + tp) * 75 + f] * tW[tp * TT + t];
    out16[i] = (_Float16)s;
}

// ---------------- concat along feature dim: writes f32 and/or f16 -----------
__global__ void concat_kernel(const float* __restrict__ a, const float* __restrict__ b,
                              float* __restrict__ o, _Float16* __restrict__ o16, int d)
{
    const long i = (long)blockIdx.x * blockDim.x + threadIdx.x;
    if (i >= NROWS * (long)(2 * d)) return;
    const long row = i / (2 * d);
    const int  c   = (int)(i % (2 * d));
    const float v = (c < d) ? a[row * d + c] : b[row * d + (c - d)];
    if (o)   o[i] = v;
    if (o16) o16[i] = (_Float16)v;
}

// ---------------- select last 6 time steps ----------------------------------
__global__ void select_last6(const float* __restrict__ full, float* __restrict__ out)
{
    const long i = (long)blockIdx.x * blockDim.x + threadIdx.x;
    if (i >= (long)BATCH * 6 * 75) return;
    const int f = (int)(i % 75);
    const long r = i / 75;
    const int tt = (int)(r % 6);
    const long b = r / 6;
    out[i] = full[(b * TT + 6 + tt) * 75 + f];
}

// ===========================================================================
// Host orchestration
// ===========================================================================
struct LNp  { const float *b,*g; };
struct MHAw { const _Float16 *Wq,*Wk,*Wv,*Wo; const float *bq,*bk,*bv,*bo; };
struct EncW { MHAw attn; const _Float16 *W1,*W2; const float *b1,*b2; LNp ln1,ln2; };
struct DecW { MHAw sattn,cattn; const _Float16 *W1,*W2; const float *b1,*b2; LNp ln1,ln2,ln3; };

static inline void gemm(hipStream_t st, const _Float16* A16, int lda,
                        const _Float16* WT, int kpad,
                        const float* bias, const float* res,
                        float* C, _Float16* C16, int M, int relu)
{
    dim3 g((M + BN - 1) / BN, (unsigned)(NROWS / BM));
    gemm_f16_wmma<<<g, dim3(256), 0, st>>>(A16, lda, WT, kpad, bias, res, C, C16, M, relu);
}
static inline void ln_launch(hipStream_t st, const float* in, const LNp& l,
                             float* out, _Float16* out16, int d)
{
    ln_kernel<<<dim3((unsigned)NROWS), dim3(256), 0, st>>>(in, l.g, l.b, out, out16, d);
}
static inline void attn_launch(hipStream_t st, const float* Q, const float* K,
                               const float* V, _Float16* O16, int d, int hd)
{
    attn_kernel<<<dim3(BATCH * (d / hd)), dim3(256), 0, st>>>(Q, K, V, O16, d, hd);
}

static void run_attn_block(hipStream_t st, float* S, _Float16* S16,
                           const _Float16* KV16, const MHAw& m, const LNp& l,
                           float* Qb, float* Kb, float* Vb, _Float16* O16, float* T1,
                           int d, int hd)
{
    gemm(st, S16,  d, m.Wq, d, m.bq, nullptr, Qb, nullptr, d, 0);
    gemm(st, KV16, d, m.Wk, d, m.bk, nullptr, Kb, nullptr, d, 0);
    gemm(st, KV16, d, m.Wv, d, m.bv, nullptr, Vb, nullptr, d, 0);
    attn_launch(st, Qb, Kb, Vb, O16, d, hd);
    gemm(st, O16, d, m.Wo, d, m.bo, S, T1, nullptr, d, 0);
    ln_launch(st, T1, l, S, S16, d);
}

extern "C" void kernel_launch(void* const* d_in, const int* in_sizes, int n_in,
                              void* d_out, int out_size, void* d_ws, size_t ws_size,
                              hipStream_t stream)
{
    (void)in_sizes; (void)n_in; (void)out_size; (void)ws_size;
    const float* x   = (const float*)d_in[0];
    const int*   adj = (const int*)d_in[1];

    // ---- param walk: sorted-pytree order (dict keys alphabetical) ----
    int pi = 2;
    auto P  = [&]() -> const float* { return (const float*)d_in[pi++]; };
    struct RawMHA { const float *Wk,*Wo,*Wq,*Wv,*bk,*bo,*bq,*bv; };
    auto mha = [&](RawMHA& m){ m.Wk=P(); m.Wo=P(); m.Wq=P(); m.Wv=P();
                               m.bk=P(); m.bo=P(); m.bq=P(); m.bv=P(); };
    auto ln  = [&](LNp& l){ l.b=P(); l.g=P(); };

    struct RawDec { const float *W1,*W2,*b1,*b2; RawMHA cattn; LNp ln1,ln2,ln3; RawMHA sattn; } rdec[2];
    for (int i = 0; i < 2; ++i) {
        rdec[i].W1=P(); rdec[i].W2=P(); rdec[i].b1=P(); rdec[i].b2=P();
        mha(rdec[i].cattn); ln(rdec[i].ln1); ln(rdec[i].ln2); ln(rdec[i].ln3);
        mha(rdec[i].sattn);
    }
    LNp dec_norm; ln(dec_norm);
    const float* emb_W = P(); const float* emb_b = P();
    const float* fc_W  = P(); const float* fc_b  = P();
    struct RawEnc { const float *W1,*W2; RawMHA attn; const float *b1,*b2; LNp ln1,ln2; } rsenc[2], rtenc[2];
    for (int i = 0; i < 2; ++i) {
        rsenc[i].W1=P(); rsenc[i].W2=P(); mha(rsenc[i].attn);
        rsenc[i].b1=P(); rsenc[i].b2=P(); ln(rsenc[i].ln1); ln(rsenc[i].ln2);
    }
    LNp s_norm; ln(s_norm);
    const float* spatial_W = P(); const float* spatial_a = P();
    for (int i = 0; i < 2; ++i) {
        rtenc[i].W1=P(); rtenc[i].W2=P(); mha(rtenc[i].attn);
        rtenc[i].b1=P(); rtenc[i].b2=P(); ln(rtenc[i].ln1); ln(rtenc[i].ln2);
    }
    LNp t_norm; ln(t_norm);
    const float* temp_W = P(); const float* temp_b = P();

    // ---- workspace: f32 region then f16 arena ----
    float* ws = (float*)d_ws;
    float* XS = ws;                   // N x 256
    float* XT = XS + NROWS * 256;     // N x 256
    float* Qb = XT + NROWS * 256;     // N x 512
    float* Kb = Qb + NROWS * 512;     // N x 512
    float* Vb = Kb + NROWS * 512;     // N x 512
    float* T1 = Vb + NROWS * 512;     // N x 512
    float* Yb = T1 + NROWS * 512;     // N x 512
    float* SP = Qb;                   // alias (N x 256), consumed pre-attention
    float* FC = Kb;                   // alias (N x 75), used at the very end

    _Float16* h16  = (_Float16*)(Yb + NROWS * 512);
    _Float16* X16   = h16;  h16 += NROWS * 96;
    _Float16* XTP16 = h16;  h16 += NROWS * 96;
    _Float16* XS16  = h16;  h16 += NROWS * 256;
    _Float16* XT16  = h16;  h16 += NROWS * 256;
    _Float16* X016  = h16;  h16 += NROWS * 512;
    _Float16* Yb16  = h16;  h16 += NROWS * 512;
    _Float16* O16   = h16;  h16 += NROWS * 512;
    _Float16* H16   = h16;  h16 += NROWS * 256;

    // ---- one-time weight conversion into f16 arena (bump allocator) ----
    auto cvtW = [&](const float* W, int K, int M) -> const _Float16* {
        const int Kpad = (K + 31) & ~31;
        _Float16* dst = h16; h16 += (long)M * Kpad;
        const long tot = (long)M * Kpad;
        convert_w<<<dim3((unsigned)((tot + 255) / 256)), dim3(256), 0, stream>>>(W, dst, K, M, Kpad);
        return dst;
    };
    auto cvtMHA = [&](const RawMHA& r, int d) -> MHAw {
        MHAw m;
        m.Wq = cvtW(r.Wq, d, d); m.Wk = cvtW(r.Wk, d, d);
        m.Wv = cvtW(r.Wv, d, d); m.Wo = cvtW(r.Wo, d, d);
        m.bq = r.bq; m.bk = r.bk; m.bv = r.bv; m.bo = r.bo;
        return m;
    };
    DecW dec[2];
    for (int i = 0; i < 2; ++i) {
        dec[i].sattn = cvtMHA(rdec[i].sattn, 512);
        dec[i].cattn = cvtMHA(rdec[i].cattn, 512);
        dec[i].W1 = cvtW(rdec[i].W1, 512, 256);
        dec[i].W2 = cvtW(rdec[i].W2, 256, 512);
        dec[i].b1 = rdec[i].b1; dec[i].b2 = rdec[i].b2;
        dec[i].ln1 = rdec[i].ln1; dec[i].ln2 = rdec[i].ln2; dec[i].ln3 = rdec[i].ln3;
    }
    EncW senc[2], tenc[2];
    for (int i = 0; i < 2; ++i) {
        senc[i].attn = cvtMHA(rsenc[i].attn, 256);
        senc[i].W1 = cvtW(rsenc[i].W1, 256, 256);
        senc[i].W2 = cvtW(rsenc[i].W2, 256, 256);
        senc[i].b1 = rsenc[i].b1; senc[i].b2 = rsenc[i].b2;
        senc[i].ln1 = rsenc[i].ln1; senc[i].ln2 = rsenc[i].ln2;
        tenc[i].attn = cvtMHA(rtenc[i].attn, 256);
        tenc[i].W1 = cvtW(rtenc[i].W1, 256, 256);
        tenc[i].W2 = cvtW(rtenc[i].W2, 256, 256);
        tenc[i].b1 = rtenc[i].b1; tenc[i].b2 = rtenc[i].b2;
        tenc[i].ln1 = rtenc[i].ln1; tenc[i].ln2 = rtenc[i].ln2;
    }
    const _Float16* embWT = cvtW(emb_W, 75, 256);   // Kpad 96
    const _Float16* fcWT  = cvtW(fc_W, 512, 75);

    // ---- forward pass ----
    convert_a96<<<dim3((unsigned)((NROWS * 96 + 255) / 256)), dim3(256), 0, stream>>>(x, X16);
    gat_kernel<<<dim3((unsigned)NROWS), dim3(256), 0, stream>>>(x, adj, spatial_W, spatial_a, SP);
    temporal_kernel<<<dim3((unsigned)((NROWS * 96 + 255) / 256)), dim3(256), 0, stream>>>(x, temp_W, temp_b, XTP16);
    gemm(stream, X16,   96, embWT, 96, emb_b, SP,      XS, XS16, 256, 0);
    gemm(stream, XTP16, 96, embWT, 96, emb_b, nullptr, XT, XT16, 256, 0);
    concat_kernel<<<dim3((unsigned)((NROWS * 512 + 255) / 256)), dim3(256), 0, stream>>>(XS, XT, nullptr, X016, 256);

    for (int i = 0; i < 2; ++i) {
        EncW* e = &senc[i];
        run_attn_block(stream, XS, XS16, XS16, e->attn, e->ln1, Qb, Kb, Vb, O16, T1, 256, 64);
        gemm(stream, XS16, 256, e->W1, 256, e->b1, nullptr, nullptr, H16, 256, 1);
        gemm(stream, H16,  256, e->W2, 256, e->b2, XS, T1, nullptr, 256, 0);
        ln_launch(stream, T1, e->ln2, XS, XS16, 256);
    }
    ln_launch(stream, XS, s_norm, XS, XS16, 256);
    for (int i = 0; i < 2; ++i) {
        EncW* e = &tenc[i];
        run_attn_block(stream, XT, XT16, XT16, e->attn, e->ln1, Qb, Kb, Vb, O16, T1, 256, 64);
        gemm(stream, XT16, 256, e->W1, 256, e->b1, nullptr, nullptr, H16, 256, 1);
        gemm(stream, H16,  256, e->W2, 256, e->b2, XT, T1, nullptr, 256, 0);
        ln_launch(stream, T1, e->ln2, XT, XT16, 256);
    }
    ln_launch(stream, XT, t_norm, XT, XT16, 256);

    concat_kernel<<<dim3((unsigned)((NROWS * 512 + 255) / 256)), dim3(256), 0, stream>>>(XS, XT, Yb, Yb16, 256);

    for (int i = 0; i < 2; ++i) {
        DecW* p = &dec[i];
        run_attn_block(stream, Yb, Yb16, Yb16, p->sattn, p->ln1, Qb, Kb, Vb, O16, T1, 512, 128);
        run_attn_block(stream, Yb, Yb16, X016, p->cattn, p->ln2, Qb, Kb, Vb, O16, T1, 512, 128);
        gemm(stream, Yb16, 512, p->W1, 512, p->b1, nullptr, nullptr, H16, 256, 1);
        gemm(stream, H16,  256, p->W2, 256, p->b2, Yb, T1, nullptr, 512, 0);
        ln_launch(stream, T1, p->ln3, Yb, Yb16, 512);
    }
    ln_launch(stream, Yb, dec_norm, Yb, Yb16, 512);

    gemm(stream, Yb16, 512, fcWT, 512, fc_b, nullptr, FC, nullptr, 75, 0);
    {
        const long tot = (long)BATCH * 6 * 75;
        select_last6<<<dim3((unsigned)((tot + 255) / 256)), dim3(256), 0, stream>>>(FC, (float*)d_out);
    }
}
